// LSTMRegressor_67044439491022
// MI455X (gfx1250) — matrix operand order repaired and checked by reading, not compiled, not run on hardware
//
#include <hip/hip_runtime.h>
#include <hip/hip_bf16.h>

// ---------------------------------------------------------------------------
// Persistent-LSTM for MI455X (gfx1250, wave32, WMMA).
//   B=512, T=512, IN=64, H=128, 2 layers + FC head.
// One workgroup owns a 16-row batch tile and iterates all T timesteps with
// weights resident in LDS (pre-swizzled to v_wmma B-operand layout).
// ---------------------------------------------------------------------------

typedef __attribute__((ext_vector_type(16))) _Float16 v16h;
typedef __attribute__((ext_vector_type(8)))  _Float16 v8h;
typedef __attribute__((ext_vector_type(4)))  _Float16 v4h;
typedef __attribute__((ext_vector_type(8)))  float    v8f;
typedef __attribute__((ext_vector_type(4)))  float    v4f;

#define H_     128
#define G4H    512          // 4*H gate columns
#define MTILE  16           // batch rows per workgroup
#define NB     32           // B / MTILE
#define BLOCK  256          // 8 waves

__device__ __forceinline__ float sigm(float x)      { return 1.0f / (1.0f + __expf(-x)); }
__device__ __forceinline__ float tanh_fast(float x) { return 2.0f / (1.0f + __expf(-2.0f * x)) - 1.0f; }

// ---------------------------------------------------------------------------
// LDS layout (dynamic):
//   Wl    : 32 ntiles x KC ksteps x [32 lanes x 16 halfs]  (B-operand swizzle)
//   Ab    : KC ksteps x [32 lanes x 16 halfs]              (A-operand swizzle)
//   gates : 16 x 512 f32
//   bias  : 512 f32 (b_ih + b_hh)
// A-operand 16-bit swizzle (16x32 M,K tile):
//   lane = m + 16*((r>>3)&1), half = (r&7) + 8*(r>>4), r = k & 31
// B-operand 16-bit swizzle (32x16 K,N tile):
//   lane = n + 16*((kr>>3)&1), half = (kr&7) + 8*(kr>>4)
// ---------------------------------------------------------------------------

template<int IN_DIM, bool FIRST>
__global__ __launch_bounds__(BLOCK) void lstm_persist(
    const float*    __restrict__ x,       // FIRST: [B, T, IN_DIM] fp32
    const _Float16* __restrict__ hs_in,   // !FIRST: swizzled hs from layer 0
    const float*    __restrict__ W_ih,    // [4H, IN_DIM]
    const float*    __restrict__ W_hh,    // [4H, H]
    const float*    __restrict__ b_ih,    // [4H]
    const float*    __restrict__ b_hh,    // [4H]
    _Float16*       __restrict__ hs_out,  // FIRST: swizzled hs for layer 1
    float*          __restrict__ h_last,  // !FIRST: [B, H] fp32
    int T)
{
    constexpr int K  = IN_DIM + H_;
    constexpr int KC = K / 32;

    extern __shared__ v16h smem_v[];
    _Float16* Wl       = (_Float16*)smem_v;          // 32*KC*512 halfs
    _Float16* Ab       = Wl + 32 * KC * 512;         // KC*512 halfs
    float*    gatesLds = (float*)(Ab + KC * 512);    // 16*512 f32
    float*    biasLds  = gatesLds + MTILE * G4H;     // 512 f32

    const int tid    = threadIdx.x;
    const int lane   = tid & 31;
    const int wave   = tid >> 5;
    const int bchunk = blockIdx.x;

    // ---- one-time: stage combined [W_ih^T ; W_hh^T] weights into LDS, f16,
    //      already in the WMMA B-operand swizzle ----
    for (int e = tid * 4; e < G4H * K; e += BLOCK * 4) {
        int g  = e / K;          // gate column (N dim), 0..511
        int k0 = e % K;          // K index, 4-contiguous
        v4f wv;
        if (k0 < IN_DIM) wv = *(const v4f*)(W_ih + (size_t)g * IN_DIM + k0);
        else             wv = *(const v4f*)(W_hh + (size_t)g * H_ + (k0 - IN_DIM));
        v4h wh;
        wh[0] = (_Float16)wv[0]; wh[1] = (_Float16)wv[1];
        wh[2] = (_Float16)wv[2]; wh[3] = (_Float16)wv[3];
        int ntile = g >> 4, n = g & 15;
        int kstep = k0 >> 5, kr = k0 & 31;
        int laneb = n + (((kr >> 3) & 1) << 4);
        int hidx  = (kr & 7) + ((kr >> 4) << 3);
        *(v4h*)(Wl + (((ntile * KC + kstep) * 32 + laneb) << 4) + hidx) = wh;
    }
    // combined bias
    for (int i = tid; i < G4H; i += BLOCK) biasLds[i] = b_ih[i] + b_hh[i];
    // zero A buffer (h_{-1} = 0; x region overwritten below)
    {
        v8h z; 
        #pragma unroll
        for (int j = 0; j < 8; ++j) z[j] = (_Float16)0.0f;
        for (int i = tid; i < KC * 64; i += BLOCK) ((v8h*)Ab)[i] = z;
    }

    // stage input x_t into the A-buffer x region (k = 0 .. IN_DIM-1)
    auto load_input = [&](int t) {
        if constexpr (FIRST) {
            const int e  = tid * 4;           // 16*64 = 1024 halfs, 4/thread
            const int m  = e >> 6;
            const int d0 = e & 63;
            v4f xv = *(const v4f*)(x + ((size_t)(bchunk * MTILE + m) * T + t) * IN_DIM + d0);
            v4h xh;
            xh[0] = (_Float16)xv[0]; xh[1] = (_Float16)xv[1];
            xh[2] = (_Float16)xv[2]; xh[3] = (_Float16)xv[3];
            int kk = d0 >> 5, r = d0 & 31;
            int la = m + (((r >> 3) & 1) << 4);
            int hb = (r & 7) + ((r >> 4) << 3);
            *(v4h*)(Ab + kk * 512 + la * 16 + hb) = xh;
        } else {
            // layer-0 already stored hs in this exact swizzle: straight copy
            size_t idx8 = ((size_t)t * NB + bchunk) * 256 + tid; // 2048 halfs / 8
            ((v8h*)Ab)[tid] = ((const v8h*)hs_in)[idx8];
        }
    };
    load_input(0);
    __syncthreads();

    // per-thread persistent cell state: (m_act, h0..h0+7)
    const int m_act = tid >> 4;
    const int h0    = (tid & 15) * 8;
    float creg[8];
    #pragma unroll
    for (int j = 0; j < 8; ++j) creg[j] = 0.0f;

    for (int t = 0; t < T; ++t) {
        // ---- WMMA phase: gates[16 x 512] = A[16 x K] * W[K x 512] ----
        v8f acc[4];
        #pragma unroll
        for (int nt = 0; nt < 4; ++nt)
            #pragma unroll
            for (int j = 0; j < 8; ++j) acc[nt][j] = 0.0f;

        #pragma unroll
        for (int kk = 0; kk < KC; ++kk) {
            v16h a = ((const v16h*)Ab)[kk * 32 + lane];
            #pragma unroll
            for (int nt = 0; nt < 4; ++nt) {
                v16h bmat = ((const v16h*)Wl)[(((wave * 4 + nt) * KC + kk) * 32) + lane];
                acc[nt] = __builtin_amdgcn_wmma_f32_16x16x32_f16(
                    /*neg_a=*/false, a, /*neg_b=*/false, bmat,
                    /*c_mod=*/(short)0, acc[nt],
                    /*reuse_a=*/false, /*reuse_b=*/false);
            }
        }
        // scatter C tiles (f32 16x16, 8 VGPRs) into the gates LDS buffer
        {
            const int m0   = (lane >> 4) * 8;
            const int coll = lane & 15;
            #pragma unroll
            for (int nt = 0; nt < 4; ++nt) {
                const int col = (wave * 4 + nt) * 16 + coll;
                #pragma unroll
                for (int j = 0; j < 8; ++j)
                    gatesLds[(m0 + j) * G4H + col] = acc[nt][j];
            }
        }
        __syncthreads();

        // ---- activation phase: each thread owns 8 hidden units of one row ----
        {
            v8f gi = *(const v8f*)(gatesLds + m_act * G4H + 0 * H_ + h0);
            v8f gf = *(const v8f*)(gatesLds + m_act * G4H + 1 * H_ + h0);
            v8f gg = *(const v8f*)(gatesLds + m_act * G4H + 2 * H_ + h0);
            v8f go = *(const v8f*)(gatesLds + m_act * G4H + 3 * H_ + h0);
            v8f bi = *(const v8f*)(biasLds + 0 * H_ + h0);
            v8f bf = *(const v8f*)(biasLds + 1 * H_ + h0);
            v8f bg = *(const v8f*)(biasLds + 2 * H_ + h0);
            v8f bo = *(const v8f*)(biasLds + 3 * H_ + h0);

            v8h pkt;
            float hvals[8];
            #pragma unroll
            for (int j = 0; j < 8; ++j) {
                float iv = sigm(gi[j] + bi[j]);
                float fv = sigm(gf[j] + bf[j]);
                float gv = tanh_fast(gg[j] + bg[j]);
                float ov = sigm(go[j] + bo[j]);
                creg[j]  = fv * creg[j] + iv * gv;
                float hv = ov * tanh_fast(creg[j]);
                hvals[j] = hv;
                pkt[j]   = (_Float16)hv;
            }
            // h_t -> A-buffer h region (absolute k = IN_DIM + h0), one b128 store
            {
                int k  = IN_DIM + h0;
                int kk = k >> 5, r = k & 31;
                int la = m_act + (((r >> 3) & 1) << 4);
                int hb = (r >> 4) << 3;                   // r&7 == 0 here
                *(v8h*)(Ab + kk * 512 + la * 16 + hb) = pkt;
            }
            if constexpr (FIRST) {
                // also publish h_t to global in layer-1's A-swizzle
                int kk = h0 >> 5, r = h0 & 31;
                int la = m_act + (((r >> 3) & 1) << 4);
                int hb = (r >> 4) << 3;
                size_t idxh = ((size_t)t * NB + bchunk) * 2048 + kk * 512 + la * 16 + hb;
                ((v8h*)hs_out)[idxh >> 3] = pkt;
            } else if (t == T - 1) {
                #pragma unroll
                for (int j = 0; j < 8; ++j)
                    h_last[(size_t)(bchunk * MTILE + m_act) * H_ + h0 + j] = hvals[j];
            }
        }

        // stage x_{t+1}; prefetch x_{t+2} into L2 (global_prefetch_b8)
        if (t + 1 < T) load_input(t + 1);
        if constexpr (FIRST) {
            if (t + 2 < T)
                __builtin_prefetch(
                    x + ((size_t)(bchunk * MTILE + (tid >> 4)) * T + (t + 2)) * IN_DIM + ((tid * 4) & 63),
                    0, 3);
        }
        __syncthreads();
    }
}

// ---------------------------------------------------------------------------
// FC head: y[b] = fc2_w . relu(fc1_w @ h_last[b] + fc1_b) + fc2_b
// ---------------------------------------------------------------------------
__global__ __launch_bounds__(128) void fc_head_kernel(
    const float* __restrict__ h_last, const float* __restrict__ fc1_w,
    const float* __restrict__ fc1_b,  const float* __restrict__ fc2_w,
    const float* __restrict__ fc2_b,  float* __restrict__ out)
{
    __shared__ float red[128];
    const int b = blockIdx.x;
    const int j = threadIdx.x;
    const float* hv = h_last + (size_t)b * H_;
    float s = fc1_b[j];
    #pragma unroll 8
    for (int k = 0; k < H_; ++k) s += fc1_w[j * H_ + k] * hv[k];
    s = fmaxf(s, 0.0f) * fc2_w[j];
    red[j] = s;
    __syncthreads();
    for (int off = 64; off > 0; off >>= 1) {
        if (j < off) red[j] += red[j + off];
        __syncthreads();
    }
    if (j == 0) out[b] = red[0] + fc2_b[0];
}

// ---------------------------------------------------------------------------
extern "C" void kernel_launch(void* const* d_in, const int* in_sizes, int n_in,
                              void* d_out, int out_size, void* d_ws, size_t ws_size,
                              hipStream_t stream) {
    const float* x      = (const float*)d_in[0];   // [512, 512, 64]
    const float* W_ih0  = (const float*)d_in[1];
    const float* W_hh0  = (const float*)d_in[2];
    const float* b_ih0  = (const float*)d_in[3];
    const float* b_hh0  = (const float*)d_in[4];
    const float* W_ih1  = (const float*)d_in[5];
    const float* W_hh1  = (const float*)d_in[6];
    const float* b_ih1  = (const float*)d_in[7];
    const float* b_hh1  = (const float*)d_in[8];
    const float* fc1_w  = (const float*)d_in[9];
    const float* fc1_b  = (const float*)d_in[10];
    const float* fc2_w  = (const float*)d_in[11];
    const float* fc2_b  = (const float*)d_in[12];
    float* out = (float*)d_out;

    const int T = 512;

    // workspace: swizzled hs (T * NB * 2048 halfs = 64 MiB) + h_last fp32
    _Float16* hs     = (_Float16*)d_ws;
    float*    h_last = (float*)((char*)d_ws + (size_t)T * NB * 2048 * sizeof(_Float16));

    auto ldsBytes = [](int KC) -> size_t {
        return (size_t)(32 * KC * 512 + KC * 512) * sizeof(_Float16)
             + (size_t)(MTILE * G4H + G4H) * sizeof(float);
    };
    const size_t lds0 = ldsBytes(6);   // 232 KB  (K = 64 + 128)
    const size_t lds1 = ldsBytes(8);   // 298 KB  (K = 128 + 128)  <= 320 KB/WGP

    (void)hipFuncSetAttribute(reinterpret_cast<const void*>(&lstm_persist<64, true>),
                              hipFuncAttributeMaxDynamicSharedMemorySize, (int)lds0);
    (void)hipFuncSetAttribute(reinterpret_cast<const void*>(&lstm_persist<128, false>),
                              hipFuncAttributeMaxDynamicSharedMemorySize, (int)lds1);

    lstm_persist<64, true><<<NB, BLOCK, lds0, stream>>>(
        x, nullptr, W_ih0, W_hh0, b_ih0, b_hh0, hs, nullptr, T);
    lstm_persist<128, false><<<NB, BLOCK, lds1, stream>>>(
        nullptr, hs, W_ih1, W_hh1, b_ih1, b_hh1, nullptr, h_last, T);
    fc_head_kernel<<<512, 128, 0, stream>>>(h_last, fc1_w, fc1_b, fc2_w, fc2_b, out);
}
